// KalmanIRV2_37976100831438
// MI455X (gfx1250) — matrix operand order repaired and checked by reading, not compiled, not run on hardware
//
#include <hip/hip_runtime.h>
#include <hip/hip_bf16.h>

typedef __attribute__((ext_vector_type(16))) _Float16 v16h;
typedef __attribute__((ext_vector_type(8)))  _Float16 v8h;
typedef __attribute__((ext_vector_type(8)))  float    v8f;

#define NB    4
#define HH    64
#define WWC   64
#define L2D   4096          // HH*WWC
#define CE    96            // embed dim
#define DI    48            // d_inner
#define NST   16            // d_state
#define NPIX  16384         // NB*L2D

static __device__ __forceinline__ float sigmf(float x){ return 1.f/(1.f+__expf(-x)); }

// ---------------------------------------------------------------------------
// Weight conversion: OIHW f32 -> f16 [k][o], k = (ky*3+kx)*96 + i, o padded to OP
// ---------------------------------------------------------------------------
__global__ void wcvt_kernel(const float* __restrict__ src, _Float16* __restrict__ dst,
                            int O, int OP){
  int t = blockIdx.x*blockDim.x + threadIdx.x;
  if (t >= 864*OP) return;
  int o = t % OP, k = t / OP;
  _Float16 v = (_Float16)0.f;
  if (o < O){
    int i = k % 96, tap = k / 96;
    v = (_Float16)src[(o*96 + i)*9 + tap];
  }
  dst[t] = v;
}

// ---------------------------------------------------------------------------
// conv_first: 3 -> 96, NCHW f32 in, NHWC f16 out (small K, direct)
// ---------------------------------------------------------------------------
__global__ void conv_first_kernel(const float* __restrict__ x, const float* __restrict__ w,
                                  const float* __restrict__ bv, _Float16* __restrict__ out){
  int t = blockIdx.x*blockDim.x + threadIdx.x;
  if (t >= NPIX*96) return;
  int co = t % 96, pix = t / 96;
  int b = pix >> 12, rem = pix & 4095, h = rem >> 6, wc = rem & 63;
  float s = bv[co];
  for (int ci = 0; ci < 3; ++ci)
    for (int dy = 0; dy < 3; ++dy){
      int r = h + dy - 1; if ((unsigned)r >= 64u) continue;
      for (int dx = 0; dx < 3; ++dx){
        int c = wc + dx - 1; if ((unsigned)c >= 64u) continue;
        s += x[(b*3 + ci)*L2D + r*64 + c] * w[(co*3 + ci)*9 + dy*3 + dx];
      }
    }
  out[t] = (_Float16)s;
}

// ---------------------------------------------------------------------------
// Implicit-GEMM 3x3 conv, Cin=96, via v_wmma_f32_16x16x32_f16.
// grid = 512 (B*H*2 row-segments of 32 pixels).
// Each wave owns a 16-pixel x 32-channel output tile (2 accumulators, shared
// A fragment). Waves per block = 2 * (CoutP/32); block = 32*that.
// Weight staging: one 3x3 tap (96 K) per barrier -> 3 K-chunks x 2 WMMA each.
// op: 0=none, 1=relu, 2=sigmoid.  res: optional residual (NHWC f16, stride Cout).
// ---------------------------------------------------------------------------
__global__ void conv3x3_wmma_kernel(const _Float16* __restrict__ in,
                                    const _Float16* __restrict__ wk,
                                    const float* __restrict__ bias,
                                    const _Float16* __restrict__ res,
                                    _Float16* __restrict__ out,
                                    int CoutP, int Cout, int op){
  __shared__ _Float16 patch[3][34][96];   // rows h-1..h+1, cols w0-1..w0+32
  __shared__ _Float16 wsmT[96][96];       // [n][k_local(=c_in)] for current tap
  const int tid = threadIdx.x, NT = blockDim.x;
  const int blk = blockIdx.x;
  const int wt = blk & 1, h = (blk >> 1) & 63, b = blk >> 7;
  const int w0 = wt * 32;
  const int pixBase = b*L2D + h*64 + w0;

  // stage input patch (zero-padded borders)
  for (int t = tid; t < 3*34*96; t += NT){
    int ry = t / (34*96), r2 = t - ry*34*96;
    int cx = r2 / 96, c = r2 - cx*96;
    int row = h + ry - 1, col = w0 + cx - 1;
    _Float16 v = (_Float16)0.f;
    if ((unsigned)row < 64u && (unsigned)col < 64u)
      v = in[(b*L2D + row*64 + col)*96 + c];
    patch[ry][cx][c] = v;
  }

  const int wave = tid >> 5, lane = tid & 31;
  const int mi = wave & 1, np = wave >> 1;       // np: which 32-wide N pair
  const int hi = lane >> 4, m = lane & 15, nl = lane & 15;
  const int nb0 = np * 32;
  v8f acc0 = {}, acc1 = {};

  for (int tap = 0; tap < 9; ++tap){
    __syncthreads();
    // stage transposed weight tile for this tap: wsmT[n][c_in]
    for (int t = tid; t < 96*CoutP; t += NT){
      int c = t / CoutP, n = t - c*CoutP;
      wsmT[n][c] = wk[(tap*96 + c)*CoutP + n];
    }
    if (tap + 1 < 9) __builtin_prefetch(&wk[(tap+1)*96*CoutP], 0, 1);
    __syncthreads();

    const int ky = tap / 3, kx = tap - ky*3;
    const int cx = m + mi*16 + kx;
    const _Float16* prow = &patch[ky][cx][0];

#pragma unroll
    for (int ch = 0; ch < 3; ++ch){
      const int cb = ch * 32;
      // A fragment (16x32 f16): per-lane = two contiguous 8-half runs
      union { v16h v; v8h h2[2]; } ua;
      ua.h2[0] = *(const v8h*)(prow + cb + hi*16);
      ua.h2[1] = *(const v8h*)(prow + cb + hi*16 + 8);
      // B fragments (32x16 f16): lane n = nl, element i -> K = hi*16 + i
      v16h b0 = *(const v16h*)&wsmT[nb0 + nl][cb + hi*16];
      v16h b1 = *(const v16h*)&wsmT[nb0 + 16 + nl][cb + hi*16];
      acc0 = __builtin_amdgcn_wmma_f32_16x16x32_f16(false, ua.v, false, b0,
                                                    (short)0, acc0, false, false);
      acc1 = __builtin_amdgcn_wmma_f32_16x16x32_f16(false, ua.v, false, b1,
                                                    (short)0, acc1, false, false);
    }
  }

  // epilogue: C/D layout -> lane nl = N, VGPR j: M = j + 8*hi
  const int pixR = pixBase + mi*16;
#pragma unroll
  for (int half = 0; half < 2; ++half){
    const int co = nb0 + half*16 + nl;
    if (co < Cout){
      const float bsc = bias[co];
#pragma unroll
      for (int j = 0; j < 8; ++j){
        int pix = pixR + j + 8*hi;
        float v = (half ? acc1[j] : acc0[j]) + bsc;
        if (op == 1) v = fmaxf(v, 0.f);
        else if (op == 2) v = sigmf(v);
        if (res) v += (float)res[pix*Cout + co];
        out[pix*Cout + co] = (_Float16)v;
      }
    }
  }
}

// ---------------------------------------------------------------------------
// a = sin(feat[:, :24]), b_ = sin(feat[:, 24:])   (feat NHWC48 f16 -> NCHW f32)
// ---------------------------------------------------------------------------
__global__ void sin_split_kernel(const _Float16* __restrict__ feat,
                                 float* __restrict__ a, float* __restrict__ bo){
  int t = blockIdx.x*blockDim.x + threadIdx.x;
  if (t >= NPIX*48) return;
  int c = t % 48, pix = t / 48;
  int b = pix >> 12, l = pix & 4095;
  float v = sinf((float)feat[t]);
  if (c < 24) a[(b*24 + c)*L2D + l] = v;
  else        bo[(b*24 + (c-24))*L2D + l] = v;
}

// ---------------------------------------------------------------------------
// sigma = softmax(b).(log_softmax(b) - log_softmax(a))  per pixel over 24 ch
// ---------------------------------------------------------------------------
__global__ void kl_kernel(const float* __restrict__ a, const float* __restrict__ bb,
                          float* __restrict__ sg){
  int t = blockIdx.x*blockDim.x + threadIdx.x;
  if (t >= NPIX) return;
  int b = t >> 12, l = t & 4095;
  float la[24], lb[24], ma = -1e30f, mb = -1e30f;
#pragma unroll
  for (int c = 0; c < 24; ++c){
    la[c] = a[(b*24+c)*L2D + l];  ma = fmaxf(ma, la[c]);
    lb[c] = bb[(b*24+c)*L2D + l]; mb = fmaxf(mb, lb[c]);
  }
  float sa = 0.f, sb = 0.f;
#pragma unroll
  for (int c = 0; c < 24; ++c){ sa += __expf(la[c]-ma); sb += __expf(lb[c]-mb); }
  float lseA = ma + logf(sa), lseB = mb + logf(sb);
#pragma unroll
  for (int c = 0; c < 24; ++c){
    float lq = lb[c] - lseB;
    sg[(b*24+c)*L2D + l] = __expf(lq) * (lq - (la[c]-lseA));
  }
}

// ---------------------------------------------------------------------------
// SS2D stage kernels
// ---------------------------------------------------------------------------
__global__ void inproj_kernel(const float* __restrict__ s, const float* __restrict__ W,
                              float* __restrict__ xi_pre, float* __restrict__ z){
  int t = blockIdx.x*blockDim.x + threadIdx.x;
  if (t >= NB*96*L2D) return;
  int l = t & 4095, r = t >> 12, e = r % 96, b = r / 96;
  float acc = 0.f;
#pragma unroll
  for (int c = 0; c < 24; ++c) acc += W[e*24 + c] * s[(b*24 + c)*L2D + l];
  if (e < DI) xi_pre[(b*DI + e)*L2D + l] = acc;
  else        z[(b*DI + (e-DI))*L2D + l] = acc;
}

__global__ void dwconv_kernel(const float* __restrict__ xp, const float* __restrict__ cw,
                              const float* __restrict__ cb, float* __restrict__ xi){
  int t = blockIdx.x*blockDim.x + threadIdx.x;
  if (t >= NB*DI*L2D) return;
  int l = t & 4095, r = t >> 12, d = r % DI, b = r / DI;
  int h = l >> 6, w = l & 63;
  float s = cb[d];
  for (int dy = 0; dy < 3; ++dy){
    int rr = h + dy - 1; if ((unsigned)rr >= 64u) continue;
    for (int dx = 0; dx < 3; ++dx){
      int cc = w + dx - 1; if ((unsigned)cc >= 64u) continue;
      s += xp[(b*DI + d)*L2D + rr*64 + cc] * cw[d*9 + dy*3 + dx];
    }
  }
  xi[t] = s * sigmf(s);
}

static __device__ __forceinline__ int mapidx(int k, int l){
  int ll = (k >= 2) ? (4095 - l) : l;
  if (k & 1) ll = ((ll & 63) << 6) | (ll >> 6);
  return ll;
}

__global__ void xdbl_kernel(const float* __restrict__ xi, const float* __restrict__ xpw,
                            float* __restrict__ xdbl){
  int t = blockIdx.x*blockDim.x + threadIdx.x;
  if (t >= NB*4*34*L2D) return;
  int l = t & 4095, r = t >> 12;
  int c = r % 34; r /= 34;
  int k = r % 4,  b = r / 4;
  int li = mapidx(k, l);
  const float* wv = xpw + (k*34 + c)*DI;
  const float* xb = xi + b*DI*L2D + li;
  float acc = 0.f;
#pragma unroll
  for (int d = 0; d < DI; ++d) acc += xb[d*L2D] * wv[d];
  xdbl[((b*4 + k)*34 + c)*L2D + l] = acc;
}

__global__ void dt_kernel(const float* __restrict__ xdbl, const float* __restrict__ dtw,
                          const float* __restrict__ dtb, float* __restrict__ dto){
  int t = blockIdx.x*blockDim.x + threadIdx.x;
  if (t >= NB*4*DI*L2D) return;
  int l = t & 4095, r = t >> 12;
  int d = r % DI; r /= DI;
  int k = r % 4,  b = r / 4;
  float x0 = xdbl[((b*4+k)*34 + 0)*L2D + l];
  float x1 = xdbl[((b*4+k)*34 + 1)*L2D + l];
  float v = x0*dtw[(k*DI+d)*2+0] + x1*dtw[(k*DI+d)*2+1] + dtb[k*DI+d];
  dto[t] = (v > 20.f) ? v : log1pf(__expf(v));
}

__global__ void scan_kernel(const float* __restrict__ xi, const float* __restrict__ dtb,
                            const float* __restrict__ xdbl, const float* __restrict__ Alogs,
                            const float* __restrict__ Ds, float* __restrict__ ys){
  int t = blockIdx.x*blockDim.x + threadIdx.x;
  if (t >= NB*4*DI) return;
  int d = t % DI, r = t / DI, k = r % 4, b = r / 4;
  float A[NST], h[NST];
#pragma unroll
  for (int n = 0; n < NST; ++n){ A[n] = -__expf(Alogs[(k*DI+d)*NST + n]); h[n] = 0.f; }
  const float* xb = xi + (b*DI + d)*L2D;
  const float* dp = dtb + ((b*4+k)*DI + d)*L2D;
  const float* Bp = xdbl + ((b*4+k)*34 + 2)*L2D;
  const float* Cp = xdbl + ((b*4+k)*34 + 18)*L2D;
  float Dv = Ds[k*DI + d];
  float* yp = ys + ((b*4+k)*DI + d)*L2D;
  for (int l = 0; l < L2D; ++l){
    float u  = xb[mapidx(k, l)];
    float dt = dp[l];
    float du = dt * u;
    float y = 0.f;
#pragma unroll
    for (int n = 0; n < NST; ++n){
      float dA = __expf(dt * A[n]);
      h[n] = dA*h[n] + du*Bp[n*L2D + l];
      y += h[n]*Cp[n*L2D + l];
    }
    yp[l] = y + Dv*u;
  }
}

__global__ void combine_kernel(const float* __restrict__ ys, const float* __restrict__ z,
                               const float* __restrict__ g, const float* __restrict__ be,
                               const float* __restrict__ opw, float* __restrict__ out){
  int t = blockIdx.x*blockDim.x + threadIdx.x;
  if (t >= NPIX) return;
  int l = t & 4095, b = t >> 12;
  int lT = ((l & 63) << 6) | (l >> 6);
  float y[DI];
#pragma unroll
  for (int d = 0; d < DI; ++d){
    y[d] = ys[((b*4+0)*DI + d)*L2D + l]
         + ys[((b*4+2)*DI + d)*L2D + (4095 - l)]
         + ys[((b*4+1)*DI + d)*L2D + lT]
         + ys[((b*4+3)*DI + d)*L2D + (4095 - lT)];
  }
  float mu = 0.f;
#pragma unroll
  for (int d = 0; d < DI; ++d) mu += y[d];
  mu *= (1.f/DI);
  float var = 0.f;
#pragma unroll
  for (int d = 0; d < DI; ++d){ float dd = y[d]-mu; var += dd*dd; }
  var *= (1.f/DI);
  float rinv = rsqrtf(var + 1e-5f);
#pragma unroll
  for (int d = 0; d < DI; ++d){
    float yn = (y[d]-mu)*rinv*g[d] + be[d];
    float zz = z[(b*DI + d)*L2D + l];
    y[d] = yn * (zz * sigmf(zz));
  }
#pragma unroll
  for (int c = 0; c < 24; ++c){
    float acc = 0.f;
#pragma unroll
    for (int d = 0; d < DI; ++d) acc += y[d] * opw[c*DI + d];
    out[(b*24 + c)*L2D + l] = acc;
  }
}

// ---------------------------------------------------------------------------
// refining = a*exp(sigma_t) + bias ; refined = sin(refining)
// ---------------------------------------------------------------------------
__global__ void refine_kernel(const float* __restrict__ a, const float* __restrict__ st,
                              const float* __restrict__ bs, float* __restrict__ rf){
  int t = blockIdx.x*blockDim.x + threadIdx.x;
  if (t >= NB*24*L2D) return;
  rf[t] = sinf(a[t]*__expf(st[t]) + bs[t]);
}

// pack concat([refined,a,b_,sigma_t]) NCHW f32 -> NHWC96 f16
__global__ void pack_feat_kernel(const float* __restrict__ rf, const float* __restrict__ a,
                                 const float* __restrict__ bb, const float* __restrict__ st,
                                 _Float16* __restrict__ out){
  int t = blockIdx.x*blockDim.x + threadIdx.x;
  if (t >= NPIX*96) return;
  int c = t % 96, pix = t / 96;
  int b = pix >> 12, l = pix & 4095;
  float v;
  if      (c < 24) v = rf[(b*24 + c)*L2D + l];
  else if (c < 48) v = a [(b*24 + (c-24))*L2D + l];
  else if (c < 72) v = bb[(b*24 + (c-48))*L2D + l];
  else             v = st[(b*24 + (c-72))*L2D + l];
  out[t] = (_Float16)v;
}

// refined = refined + K*(b_ - refined)    (K NHWC24 f16)
__global__ void kupdate_kernel(float* __restrict__ rf, const _Float16* __restrict__ Kg,
                               const float* __restrict__ bb){
  int t = blockIdx.x*blockDim.x + threadIdx.x;
  if (t >= NB*24*L2D) return;
  int l = t & 4095, r = t >> 12, c = r % 24, b = r / 24;
  float v = rf[t];
  float K = (float)Kg[((b << 12) + l)*24 + c];
  rf[t] = v + K*(bb[t] - v);
}

// bin2dec over groups of 8 channels
__global__ void bin2dec_kernel(const float* __restrict__ src, float* __restrict__ dst){
  int t = blockIdx.x*blockDim.x + threadIdx.x;
  if (t >= NB*3*L2D) return;
  int l = t & 4095, r = t >> 12, cc = r % 3, b = r / 3;
  float acc = 0.f;
#pragma unroll
  for (int n = 0; n < 8; ++n)
    acc += src[(b*24 + cc*8 + n)*L2D + l] * ((float)(1 << (7 - n)) / 255.f);
  dst[t] = acc;
}

// ---------------------------------------------------------------------------
// Host side
// ---------------------------------------------------------------------------
struct SSParams {
  const float *in_proj_w, *conv_w, *conv_b, *x_proj_w, *dt_w, *dt_b,
              *A_logs, *Ds, *ln_g, *ln_b, *out_proj_w;
};

static inline int cdiv(int a, int b){ return (a + b - 1)/b; }

static void run_ss2d(const float* src, float* dst, const SSParams& P,
                     float* xi_pre, float* zbuf, float* xibuf, float* xdbl,
                     float* dtbuf, float* ysbuf, hipStream_t stream){
  inproj_kernel<<<cdiv(NB*96*L2D,256),256,0,stream>>>(src, P.in_proj_w, xi_pre, zbuf);
  dwconv_kernel<<<cdiv(NB*DI*L2D,256),256,0,stream>>>(xi_pre, P.conv_w, P.conv_b, xibuf);
  xdbl_kernel<<<cdiv(NB*4*34*L2D,256),256,0,stream>>>(xibuf, P.x_proj_w, xdbl);
  dt_kernel<<<cdiv(NB*4*DI*L2D,256),256,0,stream>>>(xdbl, P.dt_w, P.dt_b, dtbuf);
  scan_kernel<<<cdiv(NB*4*DI,256),256,0,stream>>>(xibuf, dtbuf, xdbl, P.A_logs, P.Ds, ysbuf);
  combine_kernel<<<cdiv(NPIX,256),256,0,stream>>>(ysbuf, zbuf, P.ln_g, P.ln_b, P.out_proj_w, dst);
}

extern "C" void kernel_launch(void* const* d_in, const int* in_sizes, int n_in,
                              void* d_out, int out_size, void* d_ws, size_t ws_size,
                              hipStream_t stream){
  (void)in_sizes; (void)n_in; (void)out_size; (void)ws_size;
  const float* x    = (const float*)d_in[0];
  const float* cf_w = (const float*)d_in[1];
  const float* cf_b = (const float*)d_in[2];
  const float* gw1  = (const float*)d_in[3];
  const float* gb1  = (const float*)d_in[4];
  const float* gw2  = (const float*)d_in[5];
  const float* gb2  = (const float*)d_in[6];
  const float* cb_w = (const float*)d_in[7];
  const float* cb_b = (const float*)d_in[8];
  const float* cl_w = (const float*)d_in[9];
  const float* cl_b = (const float*)d_in[10];
  SSParams ps, pb;
  auto fill = [&](SSParams& P, int base){
    P.in_proj_w = (const float*)d_in[base+0];  P.conv_w = (const float*)d_in[base+1];
    P.conv_b    = (const float*)d_in[base+2];  P.x_proj_w = (const float*)d_in[base+3];
    P.dt_w      = (const float*)d_in[base+4];  P.dt_b   = (const float*)d_in[base+5];
    P.A_logs    = (const float*)d_in[base+6];  P.Ds     = (const float*)d_in[base+7];
    P.ln_g      = (const float*)d_in[base+8];  P.ln_b   = (const float*)d_in[base+9];
    P.out_proj_w= (const float*)d_in[base+10];
  };
  fill(ps, 11); fill(pb, 22);
  const float* km_w = (const float*)d_in[33];
  const float* km_b = (const float*)d_in[34];

  float* out = (float*)d_out;
  float* bbuf = out + 2*NB*3*L2D;          // b_ output segment, also used internally

  // --- workspace carve (linear allocator, 256B aligned) ---
  char* base = (char*)d_ws; size_t off = 0;
  auto alloc = [&](size_t bytes)->void*{
    void* p = base + off; off = (off + bytes + 255) & ~(size_t)255; return p;
  };
  _Float16* wkG1 = (_Float16*)alloc((size_t)4*864*96*2);
  _Float16* wkG2 = (_Float16*)alloc((size_t)4*864*96*2);
  _Float16* wkCB = (_Float16*)alloc((size_t)864*96*2);
  _Float16* wkCL = (_Float16*)alloc((size_t)864*64*2);
  _Float16* wkKM = (_Float16*)alloc((size_t)864*32*2);
  _Float16* A0 = (_Float16*)alloc((size_t)NPIX*96*2);
  _Float16* A1 = (_Float16*)alloc((size_t)NPIX*96*2);
  _Float16* A2 = (_Float16*)alloc((size_t)NPIX*96*2);
  _Float16* A3 = (_Float16*)alloc((size_t)NPIX*96*2);
  _Float16* feat48 = (_Float16*)alloc((size_t)NPIX*48*2);
  float* aBuf   = (float*)alloc((size_t)NB*24*L2D*4);
  float* sigma  = (float*)alloc((size_t)NB*24*L2D*4);
  float* sigmaT = (float*)alloc((size_t)NB*24*L2D*4);
  float* biasO  = (float*)alloc((size_t)NB*24*L2D*4);
  float* refined= (float*)alloc((size_t)NB*24*L2D*4);
  _Float16* featK = (_Float16*)alloc((size_t)NPIX*96*2);
  _Float16* Kg    = (_Float16*)alloc((size_t)NPIX*24*2);
  float* xi_pre = (float*)alloc((size_t)NB*DI*L2D*4);
  float* zbuf   = (float*)alloc((size_t)NB*DI*L2D*4);
  float* xibuf  = (float*)alloc((size_t)NB*DI*L2D*4);
  float* xdbl   = (float*)alloc((size_t)NB*4*34*L2D*4);
  float* dtbuf  = (float*)alloc((size_t)NB*4*DI*L2D*4);
  float* ysbuf  = (float*)alloc((size_t)NB*4*DI*L2D*4);

  // --- weight conversion to WMMA-friendly f16 [k][o] ---
  for (int g = 0; g < 4; ++g){
    wcvt_kernel<<<cdiv(864*96,256),256,0,stream>>>(gw1 + (size_t)g*96*96*9, wkG1 + (size_t)g*864*96, 96, 96);
    wcvt_kernel<<<cdiv(864*96,256),256,0,stream>>>(gw2 + (size_t)g*96*96*9, wkG2 + (size_t)g*864*96, 96, 96);
  }
  wcvt_kernel<<<cdiv(864*96,256),256,0,stream>>>(cb_w, wkCB, 96, 96);
  wcvt_kernel<<<cdiv(864*64,256),256,0,stream>>>(cl_w, wkCL, 48, 64);
  wcvt_kernel<<<cdiv(864*32,256),256,0,stream>>>(km_w, wkKM, 24, 32);

  // --- IRNet ---
  conv_first_kernel<<<cdiv(NPIX*96,256),256,0,stream>>>(x, cf_w, cf_b, A0);
  const dim3 cgrid(512);
  _Float16* pp[2] = {A1, A3};
  _Float16* body = A0;
  for (int g = 0; g < 4; ++g){
    conv3x3_wmma_kernel<<<cgrid, 192, 0, stream>>>(body, wkG1 + (size_t)g*864*96, gb1 + g*96,
                                                   nullptr, A2, 96, 96, 1);
    _Float16* nb = pp[g & 1];
    conv3x3_wmma_kernel<<<cgrid, 192, 0, stream>>>(A2, wkG2 + (size_t)g*864*96, gb2 + g*96,
                                                   body, nb, 96, 96, 0);
    body = nb;
  }
  conv3x3_wmma_kernel<<<cgrid, 192, 0, stream>>>(body, wkCB, cb_b, A0, A1, 96, 96, 0);
  conv3x3_wmma_kernel<<<cgrid, 128, 0, stream>>>(A1, wkCL, cl_b, nullptr, feat48, 64, 48, 0);

  // --- a, b_, sigma ---
  sin_split_kernel<<<cdiv(NPIX*48,256),256,0,stream>>>(feat48, aBuf, bbuf);
  kl_kernel<<<cdiv(NPIX,256),256,0,stream>>>(aBuf, bbuf, sigma);

  // --- SS2D x2 ---
  run_ss2d(sigma, sigmaT, ps, xi_pre, zbuf, xibuf, xdbl, dtbuf, ysbuf, stream);
  run_ss2d(aBuf,  biasO,  pb, xi_pre, zbuf, xibuf, xdbl, dtbuf, ysbuf, stream);

  // --- refine + kalman ---
  refine_kernel<<<cdiv(NB*24*L2D,256),256,0,stream>>>(aBuf, sigmaT, biasO, refined);
  pack_feat_kernel<<<cdiv(NPIX*96,256),256,0,stream>>>(refined, aBuf, bbuf, sigmaT, featK);
  conv3x3_wmma_kernel<<<cgrid, 64, 0, stream>>>(featK, wkKM, km_b, nullptr, Kg, 32, 24, 2);
  kupdate_kernel<<<cdiv(NB*24*L2D,256),256,0,stream>>>(refined, Kg, bbuf);

  // --- outputs ---
  bin2dec_kernel<<<cdiv(NB*3*L2D,256),256,0,stream>>>(refined, out);
  bin2dec_kernel<<<cdiv(NB*3*L2D,256),256,0,stream>>>(aBuf, out + NB*3*L2D);
  // b_ already written directly into out + 2*NB*3*L2D by sin_split_kernel
}
